// TreeLSTM_33200097198899
// MI455X (gfx1250) — matrix operand order repaired and checked by reading, not compiled, not run on hardware
//
#include <hip/hip_runtime.h>
#include <hip/hip_bf16.h>
#include <math.h>

typedef unsigned short u16;
typedef __bf16 bf16_t;
typedef bf16_t v16bf __attribute__((ext_vector_type(16)));
typedef float  v8f   __attribute__((ext_vector_type(8)));

#define BATCH  64
#define LEAVES 2048
#define XDIM   300
#define XPAD   320
#define HDIM   256
#define NCLS   5
#define TOTAL_NODES (BATCH * (2 * LEAVES - 1))   // 262080

union AFrag {
    v16bf v;
    u16   s[16];
    uint4 q[2];
};

__device__ __forceinline__ u16 f2bf(float f) {
    union { float f; unsigned u; } x; x.f = f;
    unsigned r = x.u + 0x7FFFu + ((x.u >> 16) & 1u);   // round-to-nearest-even
    return (u16)(r >> 16);
}

__device__ __forceinline__ float sigmoidf_(float x) {
    return 1.0f / (1.0f + __expf(-x));
}

// Swizzled B-matrix fragment record: for tile (nt, kt), 32 lanes x 16 bf16
// contiguous.  Lane l covers n = nt*16 + (l&15), k = kt*32 + (l>>4)*16 + e.
__device__ __forceinline__ const v16bf* frag_ptr(const u16* w, int ktiles,
                                                 int nt, int kt, int lane) {
    return (const v16bf*)(w + (((size_t)(nt * ktiles + kt) * 32 + lane) * 16));
}

// Convert fp32 row-major [Ksrc x N] weights -> zero-padded [Kpad x N] bf16 in
// WMMA B-fragment order (one 32B contiguous fragment per lane per k-step).
__global__ void convert_swizzle_kernel(const float* __restrict__ src,
                                       u16* __restrict__ dst,
                                       int Kpad, int Ksrc, int N) {
    int idx = blockIdx.x * blockDim.x + threadIdx.x;
    int total = Kpad * N;
    if (idx >= total) return;
    int e    = idx & 15;
    int rec  = idx >> 4;
    int lane = rec & 31;
    int tile = rec >> 5;
    int ktiles = Kpad >> 5;
    int kt = tile % ktiles;
    int nt = tile / ktiles;
    int k = kt * 32 + ((lane >> 4) << 4) + e;
    int n = nt * 16 + (lane & 15);
    float v = (k < Ksrc) ? src[(size_t)k * N + n] : 0.0f;
    dst[idx] = f2bf(v);
}

// One 16x16 output tile: A from LDS (bf16, contiguous k = 16*(lane>>4) + e per lane), B from
// swizzled global weights.  K = ktiles*32, f32 accumulate via v_wmma.
template <int LDK>
__device__ __forceinline__ v8f wmma_tile(const u16 (*Atile)[LDK],
                                         const u16* __restrict__ wfrag,
                                         int ktiles, int nt, int lane) {
    v8f acc = {0.f, 0.f, 0.f, 0.f, 0.f, 0.f, 0.f, 0.f};
    const int m  = lane & 15;
    const int kb = (lane >> 4) * 16;     // A layout: lanes 16-31 start at K=16
    for (int kt = 0; kt < ktiles; ++kt) {
        AFrag a;
        a.q[0] = *(const uint4*)&Atile[m][kt * 32 + kb];        // K .. K+7
        a.q[1] = *(const uint4*)&Atile[m][kt * 32 + 8 + kb];    // K+8 .. K+15
        AFrag b;
        b.v = *frag_ptr(wfrag, ktiles, nt, kt, lane);
        acc = __builtin_amdgcn_wmma_f32_16x16x32_bf16(
                  false, a.v, false, b.v, (short)0, acc, false, false);
    }
    return acc;
}

// ---------------- Leaf level: iou = embed(word) @ W_iou + b; gates, c_in = 0.
__global__ __launch_bounds__(256, 2)
void tree_leaf_kernel(const int* __restrict__ wordid,
                      const float* __restrict__ emb,
                      const u16* __restrict__ wiou_sw,
                      const float* __restrict__ b_wiou,
                      float* __restrict__ h_all,
                      float* __restrict__ c_out) {
    __shared__ u16   Atile[16][XPAD];       // 10 KB
    __shared__ float iou[16][3 * HDIM];     // 48 KB
    const int tid = threadIdx.x;
    const int rowBase = blockIdx.x * 16;

    // Stage gathered embedding rows as bf16 (K padded 300 -> 320 with zeros).
    for (int idx = tid; idx < 16 * XPAD; idx += 256) {
        int r = idx / XPAD, k = idx % XPAD;
        int g = rowBase + r;
        float x = 0.f;
        if (k < XDIM) x = emb[(size_t)wordid[g] * XDIM + k];
        Atile[r][k] = f2bf(x);
    }
    __syncthreads();

    const int wave = tid >> 5, lane = tid & 31;
    const int n = lane & 15, mb = (lane >> 4) * 8;   // C layout: M = mb + v
    for (int t = wave; t < 48; t += 8) {             // 768 / 16 col tiles
        v8f acc = wmma_tile<XPAD>(Atile, wiou_sw, XPAD / 32, t, lane);
        int col0 = t * 16;
        float bias = b_wiou[col0 + n];
        #pragma unroll
        for (int vv = 0; vv < 8; ++vv)
            iou[mb + vv][col0 + n] = acc[vv] + bias;
    }
    __syncthreads();

    for (int idx = tid; idx < 16 * HDIM; idx += 256) {
        int r = idx >> 8, hh = idx & 255;
        float i = iou[r][hh];
        float o = iou[r][HDIM + hh];
        float u = iou[r][2 * HDIM + hh];
        float c = sigmoidf_(i) * tanhf(u);           // c_in = 0
        float h = sigmoidf_(o) * tanhf(c);
        size_t g = (size_t)(rowBase + r) * HDIM + hh;
        h_all[g] = h;
        c_out[g] = c;
    }
}

// ---------------- Internal level: h_cat@U_f, h_cat@U_iou, f-gated c, gates.
__global__ __launch_bounds__(256, 2)
void tree_level_kernel(const float* __restrict__ h_prev,
                       const float* __restrict__ c_prev,
                       const u16* __restrict__ uf_sw,
                       const float* __restrict__ b_uf,
                       const u16* __restrict__ uiou_sw,
                       const float* __restrict__ b_uiou,
                       float* __restrict__ h_cur,
                       float* __restrict__ c_cur,
                       int n) {
    __shared__ u16   Atile[16][2 * HDIM];    // 16 KB, K = 512
    __shared__ float fbuf[16][2 * HDIM];     // 32 KB
    __shared__ float iou[16][3 * HDIM];      // 48 KB
    const int tid = threadIdx.x;
    const int rowBase = blockIdx.x * 16;

    // Stage A = concat(h_left, h_right), f32 -> bf16.
    for (int idx = tid; idx < 16 * 2 * HDIM; idx += 256) {
        int r = idx >> 9, k = idx & 511;
        int g = rowBase + r;
        int b = g / n, p = g % n;
        int child = b * 2 * n + 2 * p + (k >> 8);
        Atile[r][k] = f2bf(h_prev[(size_t)child * HDIM + (k & 255)]);
    }
    __syncthreads();

    const int wave = tid >> 5, lane = tid & 31;
    const int nn = lane & 15, mb = (lane >> 4) * 8;
    for (int t = wave; t < 80; t += 8) {     // 32 U_f tiles + 48 U_iou tiles
        if (t < 32) {
            v8f acc = wmma_tile<2 * HDIM>(Atile, uf_sw, 16, t, lane);
            int col0 = t * 16;
            float bias = b_uf[col0 + nn];
            #pragma unroll
            for (int vv = 0; vv < 8; ++vv)
                fbuf[mb + vv][col0 + nn] = acc[vv] + bias;
        } else {
            int nt = t - 32;
            v8f acc = wmma_tile<2 * HDIM>(Atile, uiou_sw, 16, nt, lane);
            int col0 = nt * 16;
            float bias = b_uiou[col0 + nn];
            #pragma unroll
            for (int vv = 0; vv < 8; ++vv)
                iou[mb + vv][col0 + nn] = acc[vv] + bias;
        }
    }
    __syncthreads();

    for (int idx = tid; idx < 16 * HDIM; idx += 256) {
        int r = idx >> 8, hh = idx & 255;
        int g = rowBase + r;
        int b = g / n, p = g % n;
        int c0 = b * 2 * n + 2 * p;
        float fl = sigmoidf_(fbuf[r][hh]);
        float fr = sigmoidf_(fbuf[r][HDIM + hh]);
        float cf = fl * c_prev[(size_t)c0 * HDIM + hh] +
                   fr * c_prev[(size_t)(c0 + 1) * HDIM + hh];
        float i = iou[r][hh];
        float o = iou[r][HDIM + hh];
        float u = iou[r][2 * HDIM + hh];
        float c = sigmoidf_(i) * tanhf(u) + cf;
        float h = sigmoidf_(o) * tanhf(c);
        h_cur[(size_t)g * HDIM + hh] = h;
        c_cur[(size_t)g * HDIM + hh] = c;
    }
}

// ---------------- Output projection: [262080 x 256] @ [256 x 5] + b.
__global__ __launch_bounds__(256)
void out_proj_kernel(const float* __restrict__ h_all,
                     const float* __restrict__ w_out,
                     const float* __restrict__ b_out,
                     float* __restrict__ out, int rows) {
    int wave = (int)((blockIdx.x * blockDim.x + threadIdx.x) >> 5);
    int lane = threadIdx.x & 31;
    if (wave >= rows) return;
    const float* hr = h_all + (size_t)wave * HDIM;
    float acc[NCLS] = {0.f, 0.f, 0.f, 0.f, 0.f};
    for (int k8 = 0; k8 < 8; ++k8) {         // 256 / 32 lanes
        int k = lane * 8 + k8;
        float hv = hr[k];
        #pragma unroll
        for (int c = 0; c < NCLS; ++c) acc[c] += hv * w_out[k * NCLS + c];
    }
    #pragma unroll
    for (int off = 16; off > 0; off >>= 1) {
        #pragma unroll
        for (int c = 0; c < NCLS; ++c) acc[c] += __shfl_down(acc[c], off, 32);
    }
    if (lane == 0) {
        #pragma unroll
        for (int c = 0; c < NCLS; ++c)
            out[(size_t)wave * NCLS + c] = acc[c] + b_out[c];
    }
}

extern "C" void kernel_launch(void* const* d_in, const int* in_sizes, int n_in,
                              void* d_out, int out_size, void* d_ws, size_t ws_size,
                              hipStream_t stream) {
    (void)in_sizes; (void)n_in; (void)out_size; (void)ws_size;
    const int*   wordid = (const int*)d_in[0];
    const float* emb    = (const float*)d_in[1];
    const float* W_iou  = (const float*)d_in[2];
    const float* b_Wiou = (const float*)d_in[3];
    const float* U_iou  = (const float*)d_in[4];
    const float* b_Uiou = (const float*)d_in[5];
    const float* U_f    = (const float*)d_in[6];
    const float* b_Uf   = (const float*)d_in[7];
    const float* W_out  = (const float*)d_in[8];
    const float* b_out  = (const float*)d_in[9];
    float* out = (float*)d_out;

    char* ws = (char*)d_ws;
    size_t off = 0;
    auto wsalloc = [&](size_t bytes) -> void* {
        void* p = ws + off;
        off = (off + bytes + 255) & ~(size_t)255;
        return p;
    };
    u16*   wiou_sw = (u16*)wsalloc((size_t)XPAD * 768 * sizeof(u16));
    u16*   uiou_sw = (u16*)wsalloc((size_t)512 * 768 * sizeof(u16));
    u16*   uf_sw   = (u16*)wsalloc((size_t)512 * 512 * sizeof(u16));
    float* h_all   = (float*)wsalloc((size_t)TOTAL_NODES * HDIM * sizeof(float));
    float* c_ping  = (float*)wsalloc((size_t)(BATCH * LEAVES) * HDIM * sizeof(float));
    float* c_pong  = (float*)wsalloc((size_t)(BATCH * LEAVES / 2) * HDIM * sizeof(float));

    // Weight conversion to fragment-swizzled bf16 (once per launch).
    {
        int tot = XPAD * 768;
        convert_swizzle_kernel<<<(tot + 255) / 256, 256, 0, stream>>>(
            W_iou, wiou_sw, XPAD, XDIM, 768);
        tot = 512 * 768;
        convert_swizzle_kernel<<<(tot + 255) / 256, 256, 0, stream>>>(
            U_iou, uiou_sw, 512, 512, 768);
        tot = 512 * 512;
        convert_swizzle_kernel<<<(tot + 255) / 256, 256, 0, stream>>>(
            U_f, uf_sw, 512, 512, 512);
    }

    // Leaves (131072 rows, 16 rows per block).
    tree_leaf_kernel<<<(BATCH * LEAVES) / 16, 256, 0, stream>>>(
        wordid, emb, wiou_sw, b_Wiou, h_all, c_ping);

    // 11 tree levels, ping-pong c buffers, level-major h_all.
    size_t prevOff = 0;
    size_t curOff  = (size_t)BATCH * LEAVES;
    float* cin  = c_ping;
    float* cout = c_pong;
    for (int n = LEAVES / 2; n >= 1; n >>= 1) {
        int rows = BATCH * n;
        tree_level_kernel<<<rows / 16, 256, 0, stream>>>(
            h_all + prevOff * HDIM, cin, uf_sw, b_Uf, uiou_sw, b_Uiou,
            h_all + curOff * HDIM, cout, n);
        prevOff = curOff;
        curOff += (size_t)rows;
        float* tmp = cin; cin = cout; cout = tmp;
    }

    // Output projection (wave per row, wave32 shuffle reduction).
    int rows = TOTAL_NODES;
    out_proj_kernel<<<(rows * 32 + 255) / 256, 256, 0, stream>>>(
        h_all, W_out, b_out, out, rows);
}
